// PatchBasedNonlocalModule_30700426232027
// MI455X (gfx1250) — compile-verified, matrix operand order checked by reading, not run on hardware
//
#include <hip/hip_runtime.h>
#include <cstdint>
#include <cstddef>

// ---------------------------------------------------------------------------
// CDNA5 (gfx1250) implementation of PatchBasedNonlocalModule.
// Convs are implicit-GEMM with v_wmma_f32_16x16x32_f16 (f16 in, f32 acc),
// double-buffered weight slabs (async-to-LDS when available), b128 LDS
// weight fragment loads, vectorized global traffic.
// ---------------------------------------------------------------------------

typedef __attribute__((ext_vector_type(16))) _Float16 v16h;
typedef __attribute__((ext_vector_type(8)))  _Float16 v8h;
typedef __attribute__((ext_vector_type(4)))  _Float16 v4h;
typedef __attribute__((ext_vector_type(8)))  float    v8f;

#define HW 256
#define NB 4
#define NPAT 4
#define PP 16 /* NPAT*NPAT */

#if defined(__has_builtin)
#if __has_builtin(__builtin_amdgcn_global_load_async_to_lds_b128) && \
    __has_builtin(__builtin_amdgcn_s_wait_asynccnt)
#define ASYNC_OK 1
#endif
#endif
#ifndef ASYNC_OK
#define ASYNC_OK 0
#endif

typedef int i32x4 __attribute__((vector_size(16)));
typedef __attribute__((address_space(1))) i32x4* gvec_ptr;  // global int4*
typedef __attribute__((address_space(3))) i32x4* lvec_ptr;  // LDS int4*

// ---------------------------------------------------------------------------
// Weight packing: fp32 OIHW [OC][IC][3][3] -> u32 pairs, chunk-transposed
// layout [K/32][OC][16] (entry = f16x2 of K=2*k2, 2*k2+1 for one oc).
// ---------------------------------------------------------------------------
__global__ __launch_bounds__(256) void pack_w3(const float* __restrict__ src,
                                               uint32_t* __restrict__ dst,
                                               int OC, int K) {
  int gid = blockIdx.x * 256 + threadIdx.x;
  int tot = OC * (K >> 1);
  if (gid >= tot) return;
  int oc = gid % OC;
  int k2 = gid / OC;
  union { _Float16 h[2]; uint32_t u; } p;
  p.h[0] = (_Float16)src[(size_t)oc * K + 2 * k2];
  p.h[1] = (_Float16)src[(size_t)oc * K + 2 * k2 + 1];
  dst[(size_t)(k2 >> 4) * OC * 16 + oc * 16 + (k2 & 15)] = p.u;
}

// p_w [P][OC=64][IC=64] fp32 -> u32 pairs [P][32][64] (k2-major, oc minor)
__global__ __launch_bounds__(256) void pack_wp(const float* __restrict__ src,
                                               uint32_t* __restrict__ dst) {
  int gid = blockIdx.x * 256 + threadIdx.x; // P*32*64 = 32768
  if (gid >= PP * 32 * 64) return;
  int p   = gid >> 11;
  int rem = gid & 2047;
  int k2  = rem >> 6;
  int oc  = rem & 63;
  union { _Float16 h[2]; uint32_t u; } q;
  q.h[0] = (_Float16)src[(size_t)p * 4096 + oc * 64 + 2 * k2];
  q.h[1] = (_Float16)src[(size_t)p * 4096 + oc * 64 + 2 * k2 + 1];
  dst[(size_t)p * 2048 + k2 * 64 + oc] = q.u;
}

// ---------------------------------------------------------------------------
// 3x3 conv, stride 1, 'same' pad, implicit GEMM with WMMA.
// MODE: 0 = single f16 src (IC ch), 1 = fp32 src, 2 = concat two 64-ch f16.
// Always ReLU.  FINAL: fp32 out = relu(conv)+xadd, else f16 (+opt residual).
// ---------------------------------------------------------------------------
template <int IC, int OC, int TW, int MODE, bool FINAL>
__global__ __launch_bounds__(256) void conv3x3_wmma(
    const float* __restrict__ in32, const _Float16* __restrict__ in0,
    const _Float16* __restrict__ in1, const uint32_t* __restrict__ wpack,
    const float* __restrict__ bias, const _Float16* __restrict__ res,
    _Float16* __restrict__ out16, float* __restrict__ out32,
    const float* __restrict__ xadd) {
  constexpr int  K    = IC * 9;
  constexpr int  NC   = K / 32;                 // K chunks
  constexpr bool VEC  = (MODE == 0);            // vector LDS stores possible
  constexpr int  PADL = VEC ? 3 : 1;            // left pad (alignment)
  constexpr int  ST   = VEC ? (TW + 8) : (TW + 4);
  constexpr int  NTX  = HW / TW;

  __shared__ uint32_t s_w[2 * OC * 20];         // padded slabs [2][OC][20]
  __shared__ _Float16 s_in[IC * 3 * ST];
  __shared__ uint16_t s_off[K];

  const int tid = threadIdx.x;
  const int bx  = blockIdx.x;
  const int tx  = bx % NTX;
  const int y   = (bx / NTX) % HW;
  const int b   = bx / (NTX * HW);
  const int x0  = tx * TW;

  // K-index -> LDS offset LUT
  for (int k = tid; k < K; k += 256) {
    int c = k / 9, rr = (k % 9) / 3, dx = k % 3;
    s_off[k] = (uint16_t)(c * 3 * ST + rr * ST + dx);
  }

  // --- stage input halo tile: interior (vectorized 4 elems) + 2 halo cols ---
  constexpr int NGI = IC * 3 * (TW / 4);
  for (int i = tid; i < NGI; i += 256) {
    int c   = i / (3 * (TW / 4));
    int rem = i % (3 * (TW / 4));
    int rr  = rem / (TW / 4);
    int j   = rem % (TW / 4);
    int ys  = y + rr - 1;
    int xg  = x0 + 4 * j;
    _Float16 h4[4] = {(_Float16)0.f, (_Float16)0.f, (_Float16)0.f, (_Float16)0.f};
    if (ys >= 0 && ys < HW) {
      if constexpr (MODE == 1) {
        float4 f = *(const float4*)&in32[(((size_t)b * IC + c) * HW + ys) * HW + xg];
        h4[0] = (_Float16)f.x; h4[1] = (_Float16)f.y;
        h4[2] = (_Float16)f.z; h4[3] = (_Float16)f.w;
      } else if constexpr (MODE == 2) {
        const _Float16* sp = (c < 64)
            ? &in0[(((size_t)b * 64 + c) * HW + ys) * HW + xg]
            : &in1[(((size_t)b * 64 + (c - 64)) * HW + ys) * HW + xg];
        v4h g = *(const v4h*)sp;
        h4[0] = g[0]; h4[1] = g[1]; h4[2] = g[2]; h4[3] = g[3];
      } else {
        v4h g = *(const v4h*)&in0[(((size_t)b * IC + c) * HW + ys) * HW + xg];
        h4[0] = g[0]; h4[1] = g[1]; h4[2] = g[2]; h4[3] = g[3];
      }
    }
    int base = c * 3 * ST + rr * ST + PADL + 1 + 4 * j;
    if constexpr (VEC) {
      v4h o; o[0] = h4[0]; o[1] = h4[1]; o[2] = h4[2]; o[3] = h4[3];
      *(v4h*)&s_in[base] = o;
    } else {
      s_in[base + 0] = h4[0]; s_in[base + 1] = h4[1];
      s_in[base + 2] = h4[2]; s_in[base + 3] = h4[3];
    }
  }
  for (int i = tid; i < IC * 3 * 2; i += 256) {
    int c = i / 6, rem = i % 6, rr = rem >> 1, side = rem & 1;
    int ys   = y + rr - 1;
    int colv = side ? (TW + 1) : 0;
    int xg   = x0 + colv - 1;
    _Float16 v = (_Float16)0.f;
    if (ys >= 0 && ys < HW && xg >= 0 && xg < HW) {
      if constexpr (MODE == 1)
        v = (_Float16)in32[(((size_t)b * IC + c) * HW + ys) * HW + xg];
      else if constexpr (MODE == 2)
        v = (c < 64) ? in0[(((size_t)b * 64 + c) * HW + ys) * HW + xg]
                     : in1[(((size_t)b * 64 + (c - 64)) * HW + ys) * HW + xg];
      else
        v = in0[(((size_t)b * IC + c) * HW + ys) * HW + xg];
    }
    s_in[c * 3 * ST + rr * ST + PADL + colv] = v;
  }

  // --- wave/tile mapping ---
  constexpr int PT_TOT = TW / 16;
  constexpr int NT_TOT = OC / 16;
  constexpr int NPW    = (NT_TOT * PT_TOT) / 8;
  const int wid   = tid >> 5;
  const int lane  = tid & 31;
  const int pt    = wid % PT_TOT;
  const int nbase = (wid / PT_TOT) * NPW;
  const int M     = lane & 15;
  const int half  = lane >> 4;
  const int px    = pt * 16 + M + PADL;

  int kb[8];
#pragma unroll
  for (int v = 0; v < 8; ++v)
    kb[v] = (v < 4) ? (2 * v + 8 * half) : (16 + 2 * (v - 4) + 8 * half);

  v8f acc[NPW];
#pragma unroll
  for (int n = 0; n < NPW; ++n)
#pragma unroll
    for (int j = 0; j < 8; ++j) acc[n][j] = 0.0f;

  // --- weight slab pipeline: [chunk][oc][16] global -> [oc][20] LDS ---
  constexpr int LPT = (16 * OC) / 256;  // u32 per thread per chunk (4 or 8)
  constexpr int NG4 = LPT / 4;          // b128 groups per thread
  const int f0 = tid * LPT;

#if ASYNC_OK
  auto issue = [&](int kc, int buf) {
    const uint32_t* g = wpack + (size_t)kc * 16 * OC + f0;
#pragma unroll
    for (int j = 0; j < NG4; ++j) {
      int f = f0 + 4 * j;
      int o = (f >> 4) * 20 + (f & 15);
      __builtin_amdgcn_global_load_async_to_lds_b128(
          (gvec_ptr)(g + 4 * j), (lvec_ptr)(&s_w[buf * OC * 20 + o]), 0, 0);
    }
  };
  issue(0, 0);
#else
  uint4 wreg[NG4];
  auto gload = [&](int kc) {
    const uint32_t* g = wpack + (size_t)kc * 16 * OC + f0;
#pragma unroll
    for (int j = 0; j < NG4; ++j) wreg[j] = *(const uint4*)(g + 4 * j);
  };
  auto sstore = [&](int buf) {
#pragma unroll
    for (int j = 0; j < NG4; ++j) {
      int f = f0 + 4 * j;
      int o = (f >> 4) * 20 + (f & 15);
      *(uint4*)&s_w[buf * OC * 20 + o] = wreg[j];
    }
  };
  gload(0);
  sstore(0);
  if (NC > 1) gload(1);
#endif

  for (int kc = 0; kc < NC; ++kc) {
#if ASYNC_OK
    __builtin_amdgcn_s_wait_asynccnt(0);
    __syncthreads();
    if (kc + 1 < NC) issue(kc + 1, (kc + 1) & 1);
#else
    __syncthreads();
    if (kc + 1 < NC) {
      sstore((kc + 1) & 1);
      if (kc + 2 < NC) gload(kc + 2);
    }
#endif
    const uint32_t* wbase = s_w + (kc & 1) * (OC * 20);

    union AU { v16h v; _Float16 e[16]; } au;
#pragma unroll
    for (int v = 0; v < 8; ++v) {
      int k0 = kc * 32 + kb[v];
      au.e[2 * v]     = s_in[s_off[k0] + px];
      au.e[2 * v + 1] = s_in[s_off[k0 + 1] + px];
    }
#pragma unroll
    for (int nt = 0; nt < NPW; ++nt) {
      int col = (nbase + nt) * 16 + M;
      const uint32_t* wr = wbase + col * 20;
      uint4 l0 = *(const uint4*)(wr + 4 * half);
      uint4 l1 = *(const uint4*)(wr + 8 + 4 * half);
      union BU { v16h v; uint32_t u[8]; } bu;
      bu.u[0] = l0.x; bu.u[1] = l0.y; bu.u[2] = l0.z; bu.u[3] = l0.w;
      bu.u[4] = l1.x; bu.u[5] = l1.y; bu.u[6] = l1.z; bu.u[7] = l1.w;
      acc[nt] = __builtin_amdgcn_wmma_f32_16x16x32_f16(
          false, au.v, false, bu.v, (short)0, acc[nt], false, false);
    }
  }

  // --- epilogue (vectorized stores; 8 consecutive pixels per row-group) ---
#pragma unroll
  for (int nt = 0; nt < NPW; ++nt) {
    int ch = (nbase + nt) * 16 + M;
    float bv = bias[ch];
    int xb = x0 + pt * 16 + 8 * half;
    size_t obase = (((size_t)b * OC + ch) * HW + y) * HW + xb;
    if constexpr (FINAL) {
      v8f xv = *(const v8f*)&xadd[obase];
      v8f o;
#pragma unroll
      for (int r = 0; r < 8; ++r) o[r] = fmaxf(acc[nt][r] + bv, 0.0f) + xv[r];
      *(v8f*)&out32[obase] = o;
    } else {
      v8h o;
      if (res) {
        v8h rv = *(const v8h*)&res[obase];
#pragma unroll
        for (int r = 0; r < 8; ++r)
          o[r] = (_Float16)(fmaxf(acc[nt][r] + bv, 0.0f) + (float)rv[r]);
      } else {
#pragma unroll
        for (int r = 0; r < 8; ++r)
          o[r] = (_Float16)fmaxf(acc[nt][r] + bv, 0.0f);
      }
      *(v8h*)&out16[obase] = o;
    }
  }
}

// ---------------------------------------------------------------------------
// Per-patch 1x1 conv (64x64 GEMM per patch), no ReLU.  TW = 128 pixels.
// ---------------------------------------------------------------------------
__global__ __launch_bounds__(256) void patch_conv_wmma(
    const _Float16* __restrict__ in, const uint32_t* __restrict__ wpack,
    const float* __restrict__ bias, _Float16* __restrict__ out) {
  __shared__ _Float16 s_in[64 * 128];
  __shared__ uint32_t s_w[2 * 2048];

  const int tid = threadIdx.x;
  const int bx  = blockIdx.x;
  const int tx  = bx % 2;
  const int y   = (bx / 2) % HW;
  const int b   = bx / (2 * HW);
  const int x0  = tx * 128;
  const int pL  = (y >> 6) * NPAT + (x0 >> 6);

  for (int i = tid; i < 64 * 16; i += 256) {  // 8 halves per group
    int c = i >> 4, j = i & 15;
    *(v8h*)&s_in[c * 128 + 8 * j] =
        *(const v8h*)&in[(((size_t)b * 64 + c) * HW + y) * HW + x0 + 8 * j];
  }
  for (int i = tid; i < 4096; i += 256) {
    int side = i >> 11, r = i & 2047;
    s_w[i] = wpack[(size_t)(pL + side) * 2048 + r];
  }
  __syncthreads();

  const int wid  = tid >> 5;
  const int lane = tid & 31;
  const int pt   = wid;
  const int side = pt >> 2;
  const int M    = lane & 15;
  const int half = lane >> 4;
  const int px   = pt * 16 + M;

  int kb[8];
#pragma unroll
  for (int v = 0; v < 8; ++v)
    kb[v] = (v < 4) ? (2 * v + 8 * half) : (16 + 2 * (v - 4) + 8 * half);

  v8f acc[4];
#pragma unroll
  for (int n = 0; n < 4; ++n)
#pragma unroll
    for (int j = 0; j < 8; ++j) acc[n][j] = 0.0f;

#pragma unroll
  for (int kc = 0; kc < 2; ++kc) {
    union AU { v16h v; _Float16 e[16]; } au;
#pragma unroll
    for (int v = 0; v < 8; ++v) {
      int k0 = kc * 32 + kb[v];
      au.e[2 * v]     = s_in[k0 * 128 + px];
      au.e[2 * v + 1] = s_in[(k0 + 1) * 128 + px];
    }
#pragma unroll
    for (int nt = 0; nt < 4; ++nt) {
      union BU { v16h v; uint32_t u[8]; } bu;
#pragma unroll
      for (int v = 0; v < 8; ++v)
        bu.u[v] = s_w[side * 2048 + ((kc * 32 + kb[v]) >> 1) * 64 + nt * 16 + M];
      acc[nt] = __builtin_amdgcn_wmma_f32_16x16x32_f16(
          false, au.v, false, bu.v, (short)0, acc[nt], false, false);
    }
  }

  const int patch = pL + side;
#pragma unroll
  for (int nt = 0; nt < 4; ++nt) {
    int ch = nt * 16 + M;
    float bv = bias[patch * 64 + ch];
    int xb = x0 + pt * 16 + 8 * half;
    v8h o;
#pragma unroll
    for (int r = 0; r < 8; ++r) o[r] = (_Float16)(acc[nt][r] + bv);
    *(v8h*)&out[(((size_t)b * 64 + ch) * HW + y) * HW + xb] = o;
  }
}

// ---------------------------------------------------------------------------
// Reductions for pairwise patch distances
// ---------------------------------------------------------------------------
__device__ inline float block_reduce256(float v, float* sm) {
  int tid = threadIdx.x;
  sm[tid] = v;
  __syncthreads();
  for (int s = 128; s > 0; s >>= 1) {
    if (tid < s) sm[tid] += sm[tid + s];
    __syncthreads();
  }
  return sm[0];
}

__global__ __launch_bounds__(256) void patch_norms(
    const _Float16* __restrict__ f1, const _Float16* __restrict__ f2,
    float* __restrict__ norms) {
  __shared__ float sm[256];
  const int which = blockIdx.y;
  const _Float16* src = which ? f2 : f1;
  const int idx = blockIdx.x;  // b*16 + p
  const int b = idx >> 4, p = idx & 15;
  const int py = (p >> 2) * 64, pxx = (p & 3) * 64;
  float acc = 0.0f;
  for (int i = threadIdx.x; i < 64 * 64 * 8; i += 256) {
    int c = i >> 9, dy = (i >> 3) & 63, d8 = i & 7;
    v8h g = *(const v8h*)&src[(((size_t)b * 64 + c) * HW + py + dy) * HW + pxx + 8 * d8];
#pragma unroll
    for (int j = 0; j < 8; ++j) { float v = (float)g[j]; acc += v * v; }
  }
  float t = block_reduce256(acc, sm);
  if (threadIdx.x == 0) norms[which * (NB * PP) + idx] = t;
}

__global__ __launch_bounds__(256) void patch_cross(
    const _Float16* __restrict__ f1, const _Float16* __restrict__ f2,
    float* __restrict__ cross) {
  __shared__ float sm[256];
  const int idx = blockIdx.x;  // b*256 + p*16 + q
  const int b = idx >> 8, p = (idx >> 4) & 15, q = idx & 15;
  const int py = (p >> 2) * 64, pxx = (p & 3) * 64;
  const int qy = (q >> 2) * 64, qxx = (q & 3) * 64;
  float acc = 0.0f;
  for (int i = threadIdx.x; i < 64 * 64 * 8; i += 256) {
    int c = i >> 9, dy = (i >> 3) & 63, d8 = i & 7;
    v8h a = *(const v8h*)&f1[(((size_t)b * 64 + c) * HW + py + dy) * HW + pxx + 8 * d8];
    v8h g = *(const v8h*)&f2[(((size_t)b * 64 + c) * HW + qy + dy) * HW + qxx + 8 * d8];
#pragma unroll
    for (int j = 0; j < 8; ++j) acc += (float)a[j] * (float)g[j];
  }
  float t = block_reduce256(acc, sm);
  if (threadIdx.x == 0) cross[idx] = t;
}

__global__ __launch_bounds__(64) void attn_softmax(
    const float* __restrict__ norms, const float* __restrict__ cross,
    float* __restrict__ attn) {
  int t = threadIdx.x;
  if (t >= NB * PP) return;
  int b = t >> 4, p = t & 15;
  float n1 = norms[b * 16 + p];
  float z[16];
  float mx = -1e30f;
#pragma unroll
  for (int q = 0; q < 16; ++q) {
    float d = n1 + norms[NB * PP + b * 16 + q] - 2.0f * cross[b * 256 + p * 16 + q];
    if (p == q) d = 1e8f;
    z[q] = expf(-d);
    mx = fmaxf(mx, z[q]);
  }
  float s = 0.0f;
#pragma unroll
  for (int q = 0; q < 16; ++q) { z[q] = expf(z[q] - mx); s += z[q]; }
  float inv = 1.0f / s;
#pragma unroll
  for (int q = 0; q < 16; ++q) attn[b * 256 + p * 16 + q] = z[q] * inv;
}

// out_mid = sum_q attn[b,p(y,x),q] * x2p[q-region]; 8 outputs per thread
__global__ __launch_bounds__(256) void attn_combine(
    const float* __restrict__ attn, const _Float16* __restrict__ x2p,
    _Float16* __restrict__ out) {
  int t = blockIdx.x * 256 + threadIdx.x;  // 2^21 groups of 8
  int x  = (t & 31) * 8;
  int y  = (t >> 5) & 255;
  int c  = (t >> 13) & 63;
  int b  = t >> 19;
  int p  = (y >> 6) * NPAT + (x >> 6);
  int dy = y & 63, dxb = x & 63;
  float a8[8];
#pragma unroll
  for (int j = 0; j < 8; ++j) a8[j] = 0.0f;
#pragma unroll
  for (int q = 0; q < 16; ++q) {
    float w = attn[b * 256 + p * 16 + q];
    v8h g = *(const v8h*)&x2p[(((size_t)b * 64 + c) * HW + (q >> 2) * 64 + dy) * HW +
                              (q & 3) * 64 + dxb];
#pragma unroll
    for (int j = 0; j < 8; ++j) a8[j] += w * (float)g[j];
  }
  v8h o;
#pragma unroll
  for (int j = 0; j < 8; ++j) o[j] = (_Float16)a8[j];
  *(v8h*)&out[(size_t)t * 8] = o;
}

// ---------------------------------------------------------------------------
// Launch
// ---------------------------------------------------------------------------
extern "C" void kernel_launch(void* const* d_in, const int* in_sizes, int n_in,
                              void* d_out, int out_size, void* d_ws,
                              size_t ws_size, hipStream_t stream) {
  const float* x     = (const float*)d_in[0];
  const float* pre_w = (const float*)d_in[1];
  const float* pre_b = (const float*)d_in[2];
  const float* c1_w  = (const float*)d_in[3];
  const float* c1_b  = (const float*)d_in[4];
  const float* c2_w  = (const float*)d_in[5];
  const float* c2_b  = (const float*)d_in[6];
  const float* p1_w  = (const float*)d_in[7];
  const float* p1_b  = (const float*)d_in[8];
  const float* p2_w  = (const float*)d_in[9];
  const float* p2_b  = (const float*)d_in[10];
  const float* fin_w = (const float*)d_in[11];
  const float* fin_b = (const float*)d_in[12];

  const size_t ACT = (size_t)NB * 64 * HW * HW * sizeof(_Float16);  // 32 MiB
  char* W = (char*)d_ws;
  _Float16* act0 = (_Float16*)(W + 0 * ACT);
  _Float16* bufA = (_Float16*)(W + 1 * ACT);
  _Float16* bufB = (_Float16*)(W + 2 * ACT);
  _Float16* x1p  = (_Float16*)(W + 3 * ACT);
  size_t off = 4 * ACT;
  uint32_t* wp_pre = (uint32_t*)(W + off); off += (size_t)576 * 64 * 4;
  uint32_t* wp_c1  = (uint32_t*)(W + off); off += (size_t)3 * 288 * 64 * 4;
  uint32_t* wp_c2  = (uint32_t*)(W + off); off += (size_t)3 * 288 * 64 * 4;
  uint32_t* wp_fin = (uint32_t*)(W + off); off += (size_t)576 * 128 * 4;
  uint32_t* wp_p1  = (uint32_t*)(W + off); off += (size_t)PP * 2048 * 4;
  uint32_t* wp_p2  = (uint32_t*)(W + off); off += (size_t)PP * 2048 * 4;
  float* norms = (float*)(W + off); off += 2 * NB * PP * 4;
  float* cross = (float*)(W + off); off += NB * PP * PP * 4;
  float* attn  = (float*)(W + off); off += NB * PP * PP * 4;

  // --- pack weights ---
  pack_w3<<<144, 256, 0, stream>>>(pre_w, wp_pre, 64, 1152);
  for (int i = 0; i < 3; ++i) {
    pack_w3<<<72, 256, 0, stream>>>(c1_w + (size_t)i * 64 * 576,
                                    wp_c1 + (size_t)i * 288 * 64, 64, 576);
    pack_w3<<<72, 256, 0, stream>>>(c2_w + (size_t)i * 64 * 576,
                                    wp_c2 + (size_t)i * 288 * 64, 64, 576);
  }
  pack_w3<<<288, 256, 0, stream>>>(fin_w, wp_fin, 128, 1152);
  pack_wp<<<128, 256, 0, stream>>>(p1_w, wp_p1);
  pack_wp<<<128, 256, 0, stream>>>(p2_w, wp_p2);

  // --- x0 = relu(conv(x)) ---
  conv3x3_wmma<128, 64, 64, 1, false><<<4096, 256, 0, stream>>>(
      x, nullptr, nullptr, wp_pre, pre_b, nullptr, act0, nullptr, nullptr);

  // --- branch 1 ---
  conv3x3_wmma<64, 64, 128, 0, false><<<2048, 256, 0, stream>>>(
      nullptr, act0, nullptr, wp_c1, c1_b, nullptr, bufA, nullptr, nullptr);
  conv3x3_wmma<64, 64, 128, 0, false><<<2048, 256, 0, stream>>>(
      nullptr, bufA, nullptr, wp_c1 + 288 * 64, c1_b + 64, nullptr, bufB,
      nullptr, nullptr);
  conv3x3_wmma<64, 64, 128, 0, false><<<2048, 256, 0, stream>>>(
      nullptr, bufB, nullptr, wp_c1 + 2 * 288 * 64, c1_b + 128, act0, bufA,
      nullptr, nullptr);  // bufA = x1
  patch_conv_wmma<<<2048, 256, 0, stream>>>(bufA, wp_p1, p1_b, x1p);

  // --- branch 2 ---
  conv3x3_wmma<64, 64, 128, 0, false><<<2048, 256, 0, stream>>>(
      nullptr, act0, nullptr, wp_c2, c2_b, nullptr, bufA, nullptr, nullptr);
  conv3x3_wmma<64, 64, 128, 0, false><<<2048, 256, 0, stream>>>(
      nullptr, bufA, nullptr, wp_c2 + 288 * 64, c2_b + 64, nullptr, bufB,
      nullptr, nullptr);
  conv3x3_wmma<64, 64, 128, 0, false><<<2048, 256, 0, stream>>>(
      nullptr, bufB, nullptr, wp_c2 + 2 * 288 * 64, c2_b + 128, act0, bufA,
      nullptr, nullptr);  // bufA = x2
  patch_conv_wmma<<<2048, 256, 0, stream>>>(bufA, wp_p2, p2_b, bufB);  // x2p

  // --- distances + softmax(exp(-d)) ---
  patch_norms<<<dim3(NB * PP, 2), 256, 0, stream>>>(x1p, bufB, norms);
  patch_cross<<<NB * PP * PP, 256, 0, stream>>>(x1p, bufB, cross);
  attn_softmax<<<1, 64, 0, stream>>>(norms, cross, attn);

  // --- combine -> out_mid (bufA) ---
  attn_combine<<<8192, 256, 0, stream>>>(attn, bufB, bufA);

  // --- final: out = relu(conv(concat(out_mid, x0))) + x ---
  conv3x3_wmma<128, 128, 32, 2, true><<<8192, 256, 0, stream>>>(
      nullptr, bufA, act0, wp_fin, fin_b, nullptr, nullptr, (float*)d_out, x);
}
